// SmileLinear_35948876268234
// MI455X (gfx1250) — compile-verified, matrix operand order checked by reading, not run on hardware
//
#include <hip/hip_runtime.h>
#include <hip/hip_bf16.h>

// ---------------------------------------------------------------------------
// SmileLinear on MI455X (gfx1250, wave32, WMMA + TDM)
//
//   out = x @ shared_w^T + shared_b + sum_e w_e(t) * (x @ svh[e]^T @ u[e]^T + eb[e])
//
// All heavy math on v_wmma_f32_16x16x32_bf16:
//   GEMM-1 (small):  H2[T,384] = x_bf16 @ [svh(256 rows); gate_w(64); pad(64)]^T
//   route_scale:     tw[T,8] from H2[:,256:320]; A_ext = tw-scaled H2[:,0:256]
//   GEMM-2 (main):   C[T,OUT] = [x | A_ext] @ [shared_w | u]^T  (K = 4352)
//                    + fp32 bias epilogue
// GEMM-2 stages tiles with the Tensor Data Mover: 3 LDS buffers, 2 DMA stages
// in flight, ONE workgroup barrier per K-step, tracked with TENSORcnt.
// ---------------------------------------------------------------------------

#define T_TOK 4096   // B*S tokens
#define IND   4096
#define OUTD  4096
#define NEXP  8
#define KEXP  32
#define KRTR  8
#define KEXT  (NEXP * KEXP)   // 256 extra K columns
#define KX    (IND + KEXT)    // 4352 total K
#define NSML  384             // svh(256) + gate(64) + zero pad(64)
#define KPAD  40              // LDS row pitch (bf16): 80B = 64B data + 16B pad

typedef __attribute__((ext_vector_type(16))) __bf16 v16bf;
typedef __attribute__((ext_vector_type(8)))  float  v8f;
typedef unsigned int u32x4 __attribute__((ext_vector_type(4)));
typedef int          i32x4 __attribute__((ext_vector_type(4)));
typedef int          i32x8 __attribute__((ext_vector_type(8)));

#if defined(__has_builtin)
#if __has_builtin(__builtin_amdgcn_tensor_load_to_lds) && \
    __has_builtin(__builtin_amdgcn_s_wait_tensorcnt)
#define USE_TDM 1
#endif
#endif
#ifndef USE_TDM
#define USE_TDM 0
#endif

// Reassemble a 16-bf16 fragment from LDS with 16B-aligned loads.
__device__ __forceinline__ v16bf lds_frag(const __bf16* p) {
  union { uint4 q[2]; v16bf v; } u;
  u.q[0] = *(const uint4*)p;
  u.q[1] = *(const uint4*)(p + 8);
  return u.v;
}

// ---------------------------------------------------------------------------
// TDM: DMA a [128 rows x 32 bf16] tile (global row stride = stride_elems) into
// LDS at lds_off with 4 DWORDs of padding after every 16 DWORDs -> LDS row
// pitch 80B == KPAD*2. Issued once per wave (EXEC ignored by TDM).
// ---------------------------------------------------------------------------
#if USE_TDM
__device__ __forceinline__ void tdm_load_tile_128x32(const __bf16* gbase,
                                                     unsigned lds_off,
                                                     unsigned stride_elems,
                                                     unsigned tensor_rows) {
  unsigned long long ga = (unsigned long long)(size_t)gbase;
  // Group 0: count=1 | lds_addr | global_addr[56:0] | type=2
  u32x4 g0 = { 1u,
               lds_off,
               (unsigned)ga,
               ((unsigned)(ga >> 32) & 0x01FFFFFFu) | 0x80000000u };
  // Group 1:
  //  d0: data_size=2B (1<<16), pad_enable (1<<20), pad_interval=3 (16 DW),
  //      pad_amount=3 (4 DW)
  //  d1[31:16] = tensor_dim0[15:0]   d2[15:0] = tensor_dim0[31:16]
  //  d2[31:16] = tensor_dim1[15:0]   d3[15:0] = tensor_dim1[31:16]
  //  d3[31:16] = tile_dim0 (=32)     d4 = tile_dim1 (=128) | tile_dim2<<16 (=0)
  //  d5 = tensor_dim0_stride[31:0]   d6,d7 = hi bits / dim1_stride (=0)
  i32x8 g1 = { (int)((1u << 16) | (1u << 20) | (3u << 22) | (3u << 25)),
               (int)((stride_elems & 0xFFFFu) << 16),
               (int)(((stride_elems >> 16) & 0xFFFFu) |
                     ((tensor_rows & 0xFFFFu) << 16)),
               (int)(((tensor_rows >> 16) & 0xFFFFu) | (32u << 16)),
               128, (int)stride_elems, 0, 0 };
  i32x4 z4 = { 0, 0, 0, 0 };
#if __clang_major__ >= 23
  i32x8 z8 = { 0, 0, 0, 0, 0, 0, 0, 0 };
  __builtin_amdgcn_tensor_load_to_lds(g0, g1, z4, z4, z8, 0);
#else
  __builtin_amdgcn_tensor_load_to_lds(g0, g1, z4, z4, 0);
#endif
}
#endif

// --- convert a [rows, 4096] fp32 matrix to bf16 with row stride KX ----------
__global__ __launch_bounds__(256) void conv_row_kernel(
    const float* __restrict__ src, __bf16* __restrict__ dst, int rows) {
  size_t idx4 = ((size_t)blockIdx.x * blockDim.x + threadIdx.x) * 4;
  if (idx4 >= (size_t)rows * IND) return;
  size_t r = idx4 >> 12;
  size_t i = idx4 & (IND - 1);
  float4 v = *(const float4*)(src + idx4);
  __bf16* p = dst + r * KX + i;
  p[0] = (__bf16)v.x; p[1] = (__bf16)v.y; p[2] = (__bf16)v.z; p[3] = (__bf16)v.w;
}

// --- scatter-convert u[E, OUT, KEXP] into B columns 4096..4351 ---------------
__global__ __launch_bounds__(256) void conv_u_kernel(
    const float* __restrict__ u, __bf16* __restrict__ Bb) {
  int idx = blockIdx.x * blockDim.x + threadIdx.x;
  if (idx >= OUTD * KEXT) return;
  int o = idx >> 8;
  int j = idx & (KEXT - 1);
  int e = j >> 5, k = j & (KEXP - 1);
  Bb[(size_t)o * KX + IND + j] = (__bf16)u[((size_t)e * OUTD + o) * KEXP + k];
}

// --- build Sbuf[384, 4096] bf16 = [svh(256); gate_w(64); zeros(64)] ----------
__global__ __launch_bounds__(256) void conv_small_kernel(
    const float* __restrict__ svh, const float* __restrict__ gw,
    __bf16* __restrict__ S) {
  size_t idx4 = ((size_t)blockIdx.x * blockDim.x + threadIdx.x) * 4;
  if (idx4 >= (size_t)NSML * IND) return;
  size_t r = idx4 >> 12;
  size_t i = idx4 & (IND - 1);
  float4 v = {0.f, 0.f, 0.f, 0.f};
  if (r < 256)      v = *(const float4*)(svh + r * IND + i);
  else if (r < 320) v = *(const float4*)(gw + (r - 256) * IND + i);
  __bf16* p = S + r * IND + i;
  p[0] = (__bf16)v.x; p[1] = (__bf16)v.y; p[2] = (__bf16)v.z; p[3] = (__bf16)v.w;
}

// ---------------------------------------------------------------------------
// GEMM-1: H2[T, 384] = A(:, 0:4096; lda=KX) @ S[384,4096]^T   (WMMA, LDS-staged)
// ---------------------------------------------------------------------------
__global__ __launch_bounds__(256) void gemm_small_kernel(
    const __bf16* __restrict__ A, const __bf16* __restrict__ S,
    float* __restrict__ H2) {
  __shared__ __bf16 As[128 * KPAD];
  __shared__ __bf16 Bs[128 * KPAD];
  const int tid = threadIdx.x, lane = tid & 31, wave = tid >> 5;
  const int wm = (wave & 3) * 32, wn = (wave >> 2) * 64;
  const int m0 = blockIdx.y * 128, n0 = blockIdx.x * 128;
  const int lr = tid >> 1, lh = (tid & 1) * 16;
  const __bf16* ga = A + (size_t)(m0 + lr) * KX + lh;
  const __bf16* gb = S + (size_t)(n0 + lr) * IND + lh;
  __bf16* sa = &As[lr * KPAD + lh];
  __bf16* sv = &Bs[lr * KPAD + lh];
  const int klane = (lane >> 4) * 16, mlane = lane & 15;
  v8f acc[2][4] = {};
  for (int k0 = 0; k0 < IND; k0 += 32) {
    uint4 a0 = *(const uint4*)(ga + k0);
    uint4 a1 = *(const uint4*)(ga + k0 + 8);
    uint4 b0 = *(const uint4*)(gb + k0);
    uint4 b1 = *(const uint4*)(gb + k0 + 8);
    __builtin_prefetch((const void*)(ga + k0 + 32), 0, 0);
    __builtin_prefetch((const void*)(gb + k0 + 32), 0, 0);
    *(uint4*)sa = a0; *(uint4*)(sa + 8) = a1;
    *(uint4*)sv = b0; *(uint4*)(sv + 8) = b1;
    __syncthreads();
    v16bf af[2], bfv[4];
#pragma unroll
    for (int i = 0; i < 2; ++i)
      af[i] = lds_frag(&As[(wm + i * 16 + mlane) * KPAD + klane]);
#pragma unroll
    for (int j = 0; j < 4; ++j)
      bfv[j] = lds_frag(&Bs[(wn + j * 16 + mlane) * KPAD + klane]);
#pragma unroll
    for (int i = 0; i < 2; ++i)
#pragma unroll
      for (int j = 0; j < 4; ++j)
        acc[i][j] = __builtin_amdgcn_wmma_f32_16x16x32_bf16(
            false, af[i], false, bfv[j], (short)0, acc[i][j], false, false);
    __syncthreads();
  }
  const int rlane = (lane >> 4) * 8;
#pragma unroll
  for (int j = 0; j < 4; ++j) {
    const int col = n0 + wn + j * 16 + mlane;
#pragma unroll
    for (int i = 0; i < 2; ++i)
#pragma unroll
      for (int v = 0; v < 8; ++v) {
        const int row = m0 + wm + i * 16 + rlane + v;
        H2[(size_t)row * NSML + col] = acc[i][j][v];
      }
  }
}

// ---------------------------------------------------------------------------
// Router + scale: tw[t] from gate logits H2[t,256:320]; write scaled expert
// hidden into A's extended K columns as bf16.
// ---------------------------------------------------------------------------
__global__ __launch_bounds__(256) void route_scale_kernel(
    const float* __restrict__ H2, float* __restrict__ tw,
    __bf16* __restrict__ Abuf) {
  const int t = blockIdx.x;
  const int j = threadIdx.x;
  __shared__ float tws[NEXP];
  if (j == 0) {
    const float* gl = H2 + (size_t)t * NSML + 256;
    float nrm[NEXP];
    for (int e = 0; e < NEXP; ++e) {
      float s = 0.f;
      for (int r = 0; r < KRTR; ++r) { float g = gl[e * KRTR + r]; s += g * g; }
      nrm[e] = sqrtf(s);
    }
    float mx = -1e30f;
    for (int e = 0; e < NEXP; ++e) mx = fmaxf(mx, nrm[e]);
    float p[NEXP]; float se = 0.f;
    for (int e = 0; e < NEXP; ++e) { p[e] = __expf(nrm[e] - mx); se += p[e]; }
    for (int e = 0; e < NEXP; ++e) p[e] /= se;
    int i1 = 0;
    for (int e = 1; e < NEXP; ++e) if (p[e] > p[i1]) i1 = e;
    int i2 = (i1 == 0) ? 1 : 0;
    for (int e = 0; e < NEXP; ++e) if (e != i1 && p[e] > p[i2]) i2 = e;
    float s2 = p[i1] + p[i2];
    for (int e = 0; e < NEXP; ++e) {
      float w = (e == i1) ? p[i1] / s2 : ((e == i2) ? p[i2] / s2 : 0.f);
      tws[e] = w;
      tw[(size_t)t * NEXP + e] = w;
    }
  }
  __syncthreads();
  float w = tws[j >> 5];
  Abuf[(size_t)t * KX + IND + j] = (__bf16)(H2[(size_t)t * NSML + j] * w);
}

// ---------------------------------------------------------------------------
// GEMM-2 (main): C[T,OUT] = A[T,KX] @ B[OUT,KX]^T + biases.
// TDM pipeline: 3 LDS buffers, 2 stages in flight, 1 barrier per K-step.
// ---------------------------------------------------------------------------
__global__ __launch_bounds__(256) void gemm_main_kernel(
    const __bf16* __restrict__ A, const __bf16* __restrict__ Bm,
    const float* __restrict__ sbias, const float* __restrict__ eb,
    const float* __restrict__ tw, float* __restrict__ out) {
#if USE_TDM
  __shared__ __bf16 As[3][128 * KPAD];   // 3 x 10KB
  __shared__ __bf16 Bs[3][128 * KPAD];   // 3 x 10KB  (total 60KB)
#else
  __shared__ __bf16 As[1][128 * KPAD];
  __shared__ __bf16 Bs[1][128 * KPAD];
#endif

  const int tid = threadIdx.x, lane = tid & 31, wave = tid >> 5;
  const int wm = (wave & 3) * 32, wn = (wave >> 2) * 64;
  const int m0 = blockIdx.y * 128, n0 = blockIdx.x * 128;
  const int klane = (lane >> 4) * 16, mlane = lane & 15;
  const int NIT = KX / 32;   // 136

  v8f acc[2][4] = {};

#if USE_TDM
  const __bf16* gA = A + (size_t)m0 * KX;
  const __bf16* gB = Bm + (size_t)n0 * KX;
  if (wave == 0) {   // prologue: stages 0 and 1
    tdm_load_tile_128x32(gA,      (unsigned)(size_t)&As[0][0], KX, T_TOK);
    tdm_load_tile_128x32(gB,      (unsigned)(size_t)&Bs[0][0], KX, OUTD);
    tdm_load_tile_128x32(gA + 32, (unsigned)(size_t)&As[1][0], KX, T_TOK);
    tdm_load_tile_128x32(gB + 32, (unsigned)(size_t)&Bs[1][0], KX, OUTD);
  }
  int cur = 0;
  for (int it = 0; it < NIT; ++it) {
    if (wave == 0) {   // stage `it` landed (in-order TENSORcnt)
      if (it + 1 < NIT)
        __builtin_amdgcn_s_wait_tensorcnt(2);   // stage it+1 still in flight
      else
        __builtin_amdgcn_s_wait_tensorcnt(0);   // drain everything
    }
    __syncthreads(); // buf[cur] ready for all; buf[(cur+2)%3] free for reuse
    if (wave == 0 && it + 2 < NIT) {
      const int nb = (cur >= 1) ? cur - 1 : 2;   // (cur+2)%3
      tdm_load_tile_128x32(gA + (it + 2) * 32,
                           (unsigned)(size_t)&As[nb][0], KX, T_TOK);
      tdm_load_tile_128x32(gB + (it + 2) * 32,
                           (unsigned)(size_t)&Bs[nb][0], KX, OUTD);
    }
    v16bf af[2], bfv[4];
#pragma unroll
    for (int i = 0; i < 2; ++i)
      af[i] = lds_frag(&As[cur][(wm + i * 16 + mlane) * KPAD + klane]);
#pragma unroll
    for (int j = 0; j < 4; ++j)
      bfv[j] = lds_frag(&Bs[cur][(wn + j * 16 + mlane) * KPAD + klane]);
#pragma unroll
    for (int i = 0; i < 2; ++i)
#pragma unroll
      for (int j = 0; j < 4; ++j)
        acc[i][j] = __builtin_amdgcn_wmma_f32_16x16x32_bf16(
            false, af[i], false, bfv[j], (short)0, acc[i][j], false, false);
    cur = (cur == 2) ? 0 : cur + 1;
  }
#else
  // Fallback: manual global->LDS staging (single buffer).
  const int lr = tid >> 1, lh = (tid & 1) * 16;
  const __bf16* ga = A + (size_t)(m0 + lr) * KX + lh;
  const __bf16* gb = Bm + (size_t)(n0 + lr) * KX + lh;
  __bf16* sa = &As[0][lr * KPAD + lh];
  __bf16* sv = &Bs[0][lr * KPAD + lh];
  for (int k0 = 0; k0 < KX; k0 += 32) {
    uint4 a0 = *(const uint4*)(ga + k0);
    uint4 a1 = *(const uint4*)(ga + k0 + 8);
    uint4 b0 = *(const uint4*)(gb + k0);
    uint4 b1 = *(const uint4*)(gb + k0 + 8);
    __builtin_prefetch((const void*)(ga + k0 + 32), 0, 0);
    __builtin_prefetch((const void*)(gb + k0 + 32), 0, 0);
    *(uint4*)sa = a0; *(uint4*)(sa + 8) = a1;
    *(uint4*)sv = b0; *(uint4*)(sv + 8) = b1;
    __syncthreads();
    v16bf af[2], bfv[4];
#pragma unroll
    for (int i = 0; i < 2; ++i)
      af[i] = lds_frag(&As[0][(wm + i * 16 + mlane) * KPAD + klane]);
#pragma unroll
    for (int j = 0; j < 4; ++j)
      bfv[j] = lds_frag(&Bs[0][(wn + j * 16 + mlane) * KPAD + klane]);
#pragma unroll
    for (int i = 0; i < 2; ++i)
#pragma unroll
      for (int j = 0; j < 4; ++j)
        acc[i][j] = __builtin_amdgcn_wmma_f32_16x16x32_bf16(
            false, af[i], false, bfv[j], (short)0, acc[i][j], false, false);
    __syncthreads();
  }
#endif

  // fp32 epilogue: + shared_b[o] + sum_e tw[t][e]*eb[e][o]
  const int rlane = (lane >> 4) * 8;
#pragma unroll
  for (int j = 0; j < 4; ++j) {
    const int col = n0 + wn + j * 16 + mlane;
    const float b0 = sbias[col];
    float ebv[NEXP];
#pragma unroll
    for (int e = 0; e < NEXP; ++e) ebv[e] = eb[(size_t)e * OUTD + col];
#pragma unroll
    for (int i = 0; i < 2; ++i)
#pragma unroll
      for (int v = 0; v < 8; ++v) {
        const int row = m0 + wm + i * 16 + rlane + v;
        const float* twr = tw + (size_t)row * NEXP;
        float bias = b0;
#pragma unroll
        for (int e = 0; e < NEXP; ++e) bias += twr[e] * ebv[e];
        out[(size_t)row * OUTD + col] = acc[i][j][v] + bias;
      }
  }
}

// ---------------------------------------------------------------------------
extern "C" void kernel_launch(void* const* d_in, const int* in_sizes, int n_in,
                              void* d_out, int out_size, void* d_ws, size_t ws_size,
                              hipStream_t stream) {
  const float* x   = (const float*)d_in[0];  // [2,2048,4096]
  const float* gw  = (const float*)d_in[1];  // [64,4096]
  const float* sw  = (const float*)d_in[2];  // [4096,4096]
  const float* sb  = (const float*)d_in[3];  // [4096]
  const float* u   = (const float*)d_in[4];  // [8,4096,32]
  const float* svh = (const float*)d_in[5];  // [8,32,4096]
  const float* eb  = (const float*)d_in[6];  // [8,4096]
  float* out = (float*)d_out;

  const size_t A_BYTES = (size_t)T_TOK * KX * sizeof(__bf16);   // ~35.6 MB
  const size_t B_BYTES = (size_t)OUTD  * KX * sizeof(__bf16);   // ~35.6 MB
  const size_t S_BYTES = (size_t)NSML * IND * sizeof(__bf16);   // 3 MB
  const size_t H_BYTES = (size_t)T_TOK * NSML * sizeof(float);  // 6.3 MB
  char* ws = (char*)d_ws;
  __bf16* Abuf = (__bf16*)(ws);
  __bf16* Bbuf = (__bf16*)(ws + A_BYTES);
  __bf16* Sbuf = (__bf16*)(ws + A_BYTES + B_BYTES);
  float*  H2   = (float*)(ws + A_BYTES + B_BYTES + S_BYTES);
  float*  tw   = (float*)(ws + A_BYTES + B_BYTES + S_BYTES + H_BYTES);

  // 1) bf16 conversions / packing
  {
    int nblk = (int)(((size_t)T_TOK * IND / 4 + 255) / 256);
    conv_row_kernel<<<nblk, 256, 0, stream>>>(x, Abuf, T_TOK);
    conv_row_kernel<<<nblk, 256, 0, stream>>>(sw, Bbuf, OUTD);
    conv_u_kernel<<<(OUTD * KEXT + 255) / 256, 256, 0, stream>>>(u, Bbuf);
    int nsml = (int)(((size_t)NSML * IND / 4 + 255) / 256);
    conv_small_kernel<<<nsml, 256, 0, stream>>>(svh, gw, Sbuf);
  }
  // 2) WMMA GEMM-1: expert hiddens + gate logits in one shot
  gemm_small_kernel<<<dim3(NSML / 128, T_TOK / 128), 256, 0, stream>>>(
      Abuf, Sbuf, H2);
  // 3) router softmax/top-2 + scale expert hiddens into A's extended columns
  route_scale_kernel<<<T_TOK, 256, 0, stream>>>(H2, tw, Abuf);
  // 4) main fused WMMA GEMM (TDM pipeline)
  gemm_main_kernel<<<dim3(OUTD / 128, T_TOK / 128), 256, 0, stream>>>(
      Abuf, Bbuf, sb, eb, tw, out);
}
